// Lamm_7413113553022
// MI455X (gfx1250) — compile-verified, matrix operand order checked by reading, not run on hardware
//
#include <hip/hip_runtime.h>
#include <hip/hip_bf16.h>

typedef __attribute__((ext_vector_type(16))) _Float16 v16h;
typedef __attribute__((ext_vector_type(8)))  float    v8f;
typedef __attribute__((ext_vector_type(4)))  float    v4f;

#define BATCH_N 256

__host__ __device__ constexpr int LH_(int l){ return l==0?200 : l==1?100 : l==2?50 : 25; }
__host__ __device__ constexpr int LW_(int l){ return l==0?334 : l==1?167 : l==2?84 : 42; }
__host__ __device__ constexpr int NPIX_(int l){ return LH_(l)*LW_(l); }
__host__ __device__ constexpr int NELEM_(int l){ return BATCH_N*NPIX_(l); }      // all multiples of 16
__host__ __device__ constexpr int NCHUNK_(int l){ return NELEM_(l)/16; }
__host__ __device__ constexpr int PBLK_(int l){                                   // capped sum blocks
    int full = (NCHUNK_(l)+255)/256; return full < 512 ? full : 512;
}
__host__ __device__ constexpr int ITERS_(int l){
    int t = PBLK_(l)*256; return (NCHUNK_(l) + t - 1) / t;
}
__host__ __device__ constexpr int MBLK_(int l){ return (NPIX_(l)+255)/256; }      // mask blocks
__host__ __device__ constexpr int POFF_(int l){ int o=0; for(int i=0;i<l;++i) o+=PBLK_(i); return o; }
__host__ __device__ constexpr int MOFF_(int l){ int o=0; for(int i=0;i<l;++i) o+=MBLK_(i); return o; }
__host__ __device__ constexpr int SUM_PARTIALS_TOTAL(){ return POFF_(4); }        // 512+512+263+66 = 1353
__host__ __device__ constexpr int CNT_PARTIALS_TOTAL(){ return MOFF_(4); }        // 261+66+17+5    = 349

// ---------------------------------------------------------------------------
// Level sum via WMMA ones-trick: D = A * Ones + C. Every D column holds the
// row sums of A, so Sum(D) = 16 * Sum(A-elements); C accumulates across the
// grid-stride loop. Trip count is grid-uniform so EXEC is all-ones at every
// WMMA (ISA requirement); only the loads are chunk-guarded (zero-fill).
// Inner loop: 4x global_load_b128 + packed f32->f16 cvt + 1x v_wmma.
// ---------------------------------------------------------------------------
__global__ __launch_bounds__(256) void level_sum_kernel(const float* __restrict__ x,
                                                        int nChunks, int iters,
                                                        float* __restrict__ partials)
{
    const int tid  = threadIdx.x;
    const int gtid = blockIdx.x * 256 + tid;
    const int totalThreads = gridDim.x * 256;

    v16h ones;
    #pragma unroll
    for (int i = 0; i < 16; ++i) ones[i] = (_Float16)1.0f;

    v8f acc = {};

    int chunk = gtid;
    for (int it = 0; it < iters; ++it, chunk += totalThreads) {
        v16h a;
        if (chunk < nChunks) {
            const v4f* v = (const v4f*)(x + (size_t)chunk * 16);   // 64B aligned
            v4f d0 = v[0], d1 = v[1], d2 = v[2], d3 = v[3];
            #pragma unroll
            for (int i = 0; i < 4; ++i) {
                a[i]      = (_Float16)d0[i];
                a[4 + i]  = (_Float16)d1[i];
                a[8 + i]  = (_Float16)d2[i];
                a[12 + i] = (_Float16)d3[i];
            }
        } else {
            #pragma unroll
            for (int i = 0; i < 16; ++i) a[i] = (_Float16)0.0f;
        }
        // Re-converged: full EXEC for the matrix op.
        acc = __builtin_amdgcn_wmma_f32_16x16x32_f16(
            /*neg_a=*/false, a, /*neg_b=*/false, ones,
            /*c_mod=*/(short)0, acc, /*reuse_a=*/false, /*reuse_b=*/false);
    }

    // per-lane sum of the 8 accumulator rows
    float s = 0.0f;
    #pragma unroll
    for (int i = 0; i < 8; ++i) s += acc[i];

    // wave32 butterfly reduction (no barriers)
    #pragma unroll
    for (int off = 16; off > 0; off >>= 1) s += __shfl_xor(s, off, 32);

    __shared__ float wsum[8];
    const int wave = tid >> 5;
    if ((tid & 31) == 0) wsum[wave] = s;
    __syncthreads();
    if (tid == 0) {
        float t = 0.0f;
        #pragma unroll
        for (int i = 0; i < 8; ++i) t += wsum[i];
        partials[blockIdx.x] = t * (1.0f / 16.0f);
    }
}

// ---------------------------------------------------------------------------
// Union-of-boxes pixel count for one level. Boxes staged (clipped) in LDS.
// rintf == round-half-to-even, matching jnp.round.
// ---------------------------------------------------------------------------
__global__ __launch_bounds__(256) void mask_count_kernel(const float* __restrict__ boxes,
                                                         int nboxes,
                                                         int h, int w,
                                                         const int* __restrict__ p_imx,
                                                         const int* __restrict__ p_imy,
                                                         int* __restrict__ partials)
{
    __shared__ int bx1[256], by1[256], bx2[256], by2[256];
    const int tid = threadIdx.x;
    if (nboxes > 256) nboxes = 256;

    const float sx = (float)w / (float)(*p_imx);
    const float sy = (float)h / (float)(*p_imy);

    for (int b = tid; b < nboxes; b += 256) {
        float fx1 = boxes[b * 4 + 0] * sx;
        float fy1 = boxes[b * 4 + 1] * sy;
        float fx2 = boxes[b * 4 + 2] * sx;
        float fy2 = boxes[b * 4 + 3] * sy;
        int x1 = (int)fminf(fmaxf(rintf(fx1), 0.0f), (float)(w - 1));
        int y1 = (int)fminf(fmaxf(rintf(fy1), 0.0f), (float)(h - 1));
        int x2 = (int)fminf(fmaxf(rintf(fx2), 0.0f), (float)w);
        int y2 = (int)fminf(fmaxf(rintf(fy2), 0.0f), (float)h);
        bool valid = (x2 > x1) && (y2 > y1) && (x1 + x2 < w) && (y1 + y2 < h);
        if (!valid) { x1 = 0; x2 = 0; y1 = 0; y2 = 0; }   // empty interval
        bx1[b] = x1; bx2[b] = x2; by1[b] = y1; by2[b] = y2;
    }
    __syncthreads();

    int count = 0;
    const int total = h * w;
    for (int p = blockIdx.x * 256 + tid; p < total; p += gridDim.x * 256) {
        const int y = p / w;
        const int x = p - y * w;
        bool cov = false;
        for (int b = 0; b < nboxes; ++b)
            cov = cov || ((x >= bx1[b]) && (x < bx2[b]) && (y >= by1[b]) && (y < by2[b]));
        count += cov ? 1 : 0;
    }

    // wave32 butterfly + cross-wave combine
    #pragma unroll
    for (int off = 16; off > 0; off >>= 1) count += __shfl_xor(count, off, 32);

    __shared__ int wcnt[8];
    const int wave = tid >> 5;
    if ((tid & 31) == 0) wcnt[wave] = count;
    __syncthreads();
    if (tid == 0) {
        int t = 0;
        #pragma unroll
        for (int i = 0; i < 8; ++i) t += wcnt[i];
        partials[blockIdx.x] = t;
    }
}

// ---------------------------------------------------------------------------
// Single-block finalize: deterministic fixed-order reduction of all partials,
// loss = mean_l ( sum_l/t_l - count_l/t_l )^2 , t_l = 256*h_l*w_l.
// ---------------------------------------------------------------------------
__global__ __launch_bounds__(256) void finalize_kernel(const float* __restrict__ sumP,
                                                       const int* __restrict__ cntP,
                                                       float* __restrict__ out)
{
    __shared__ float redf[256];
    __shared__ int   redi[256];
    __shared__ float levelS[4];
    __shared__ int   levelC[4];
    const int tid = threadIdx.x;

    for (int l = 0; l < 4; ++l) {
        float s = 0.0f;
        {
            const int o = POFF_(l), c = PBLK_(l);
            for (int i = tid; i < c; i += 256) s += sumP[o + i];
        }
        redf[tid] = s;
        __syncthreads();
        #pragma unroll
        for (int off = 128; off > 0; off >>= 1) {
            if (tid < off) redf[tid] += redf[tid + off];
            __syncthreads();
        }
        if (tid == 0) levelS[l] = redf[0];
        __syncthreads();

        int ci = 0;
        {
            const int o = MOFF_(l), c = MBLK_(l);
            for (int i = tid; i < c; i += 256) ci += cntP[o + i];
        }
        redi[tid] = ci;
        __syncthreads();
        #pragma unroll
        for (int off = 128; off > 0; off >>= 1) {
            if (tid < off) redi[tid] += redi[tid + off];
            __syncthreads();
        }
        if (tid == 0) levelC[l] = redi[0];
        __syncthreads();
    }

    if (tid == 0) {
        float acc = 0.0f;
        #pragma unroll
        for (int l = 0; l < 4; ++l) {
            const float t = (float)BATCH_N * (float)NPIX_(l);
            const float d = levelS[l] / t - (float)levelC[l] / t;
            acc += d * d;
        }
        out[0] = acc * 0.25f;
    }
}

// ---------------------------------------------------------------------------
extern "C" void kernel_launch(void* const* d_in, const int* in_sizes, int n_in,
                              void* d_out, int out_size, void* d_ws, size_t ws_size,
                              hipStream_t stream) {
    (void)n_in; (void)out_size; (void)ws_size;

    const float* masks[4] = {
        (const float*)d_in[0], (const float*)d_in[1],
        (const float*)d_in[2], (const float*)d_in[3]
    };
    const float* boxes = (const float*)d_in[4];
    const int*   p_imx = (const int*)d_in[5];
    const int*   p_imy = (const int*)d_in[6];
    float*       out   = (float*)d_out;

    const int nboxes = in_sizes[4] / 4;

    float* sumP = (float*)d_ws;
    int*   cntP = (int*)((char*)d_ws + (size_t)SUM_PARTIALS_TOTAL() * sizeof(float));

    // Per-level bandwidth-bound sums (WMMA ones-trick, grid-stride accumulate)
    level_sum_kernel<<<PBLK_(0), 256, 0, stream>>>(masks[0], NCHUNK_(0), ITERS_(0), sumP + POFF_(0));
    level_sum_kernel<<<PBLK_(1), 256, 0, stream>>>(masks[1], NCHUNK_(1), ITERS_(1), sumP + POFF_(1));
    level_sum_kernel<<<PBLK_(2), 256, 0, stream>>>(masks[2], NCHUNK_(2), ITERS_(2), sumP + POFF_(2));
    level_sum_kernel<<<PBLK_(3), 256, 0, stream>>>(masks[3], NCHUNK_(3), ITERS_(3), sumP + POFF_(3));

    // Per-level union-of-boxes pixel counts
    mask_count_kernel<<<MBLK_(0), 256, 0, stream>>>(boxes, nboxes, LH_(0), LW_(0), p_imx, p_imy, cntP + MOFF_(0));
    mask_count_kernel<<<MBLK_(1), 256, 0, stream>>>(boxes, nboxes, LH_(1), LW_(1), p_imx, p_imy, cntP + MOFF_(1));
    mask_count_kernel<<<MBLK_(2), 256, 0, stream>>>(boxes, nboxes, LH_(2), LW_(2), p_imx, p_imy, cntP + MOFF_(2));
    mask_count_kernel<<<MBLK_(3), 256, 0, stream>>>(boxes, nboxes, LH_(3), LW_(3), p_imx, p_imy, cntP + MOFF_(3));

    // Deterministic final combine
    finalize_kernel<<<1, 256, 0, stream>>>(sumP, cntP, out);
}
